// CausalSelfAttention_60224031424653
// MI455X (gfx1250) — matrix operand order repaired and checked by reading, not compiled, not run on hardware
//
#include <hip/hip_runtime.h>
#include <hip/hip_fp16.h>

// ---------------------------------------------------------------------------
// CDNA5 (gfx1250) fused sliding-window attention block.
// f16 WMMA (v_wmma_f32_16x16x32_f16) with f32 accumulate; softmax in f32.
// ~155 GFLOP vs ~400MB HBM traffic (17us @ 23.3TB/s) -> matrix-core bound.
// GEMM: 32x64 tile per wave, fragments straight from global (WGP$/L2 reuse),
// prefetch ahead. Attention: 130KB LDS/WG -> 2 WGs per 320KB WGP.
// ---------------------------------------------------------------------------

typedef __attribute__((ext_vector_type(16))) _Float16 v16h;
typedef __attribute__((ext_vector_type(8)))  _Float16 v8h;
typedef __attribute__((ext_vector_type(8)))  float    v8f;

#define TLEN   4096
#define BATCH  4
#define CDIM   1024
#define NHEAD  16
#define HDIM   64
#define WIN    128   // sliding window / block size

// ---- WMMA fragment loaders (wave32, 16x16x32 f16) -------------------------
// A: 16(M) x 32(K), source row-major, leading dim ld.
// Lane l: row (l&15); K elems {kb..kb+7, kb+16..kb+23}, kb = (l>=16)*8.
__device__ __forceinline__ v16h load_frag_a(const _Float16* base, int ld, int lane) {
  const _Float16* p = base + (lane & 15) * ld + ((lane >> 4) << 3);
  v8h lo = *(const v8h*)(p);
  v8h hi = *(const v8h*)(p + 16);
  return __builtin_shufflevector(lo, hi, 0,1,2,3,4,5,6,7,8,9,10,11,12,13,14,15);
}

// B: 32(K) x 16(N); source stored as [N][K] row-major (B^T, K contiguous).
// Lane l: column (l&15); K elems {kb..kb+7, kb+16..kb+23}, kb = (l>=16)*8.
__device__ __forceinline__ v16h load_frag_b(const _Float16* base, int ld, int lane) {
  const _Float16* p = base + (lane & 15) * ld + ((lane >> 4) << 3);
  v8h lo = *(const v8h*)(p);
  v8h hi = *(const v8h*)(p + 16);
  return __builtin_shufflevector(lo, hi, 0,1,2,3,4,5,6,7,8,9,10,11,12,13,14,15);
}

__device__ __forceinline__ v8f wmma_f16(v16h a, v16h b, v8f c) {
  return __builtin_amdgcn_wmma_f32_16x16x32_f16(false, a, false, b, (short)0, c, false, false);
}

// ---- fp32 -> f16 convert, 8 elems/thread (b128 store) ---------------------
__global__ void cvt_f32_f16_v8(const float4* __restrict__ src, v8h* __restrict__ dst, long n8) {
  long i = (long)blockIdx.x * blockDim.x + threadIdx.x;
  if (i >= n8) return;
  float4 a = src[2 * i], b = src[2 * i + 1];
  v8h o;
  o[0] = (_Float16)a.x; o[1] = (_Float16)a.y; o[2] = (_Float16)a.z; o[3] = (_Float16)a.w;
  o[4] = (_Float16)b.x; o[5] = (_Float16)b.y; o[6] = (_Float16)b.z; o[7] = (_Float16)b.w;
  dst[i] = o;
}

// ---- NT GEMM: C[M,N] = A[M,K] * Bw[N,K]^T ---------------------------------
// Workgroup: 8 waves * (32x64) = 256x64 tile. Fragments direct from global;
// B fragments are shared across waves via WGP$/L2; prefetch 8 k-steps ahead.
__global__ __launch_bounds__(256)
void gemm_nt_f16(const _Float16* __restrict__ A, const _Float16* __restrict__ Bw,
                 float* __restrict__ Cf, _Float16* __restrict__ Ch,
                 int M, int N, int K, int storeHalf) {
  const int lane = threadIdx.x & 31;
  const int wave = threadIdx.x >> 5;                 // 8 waves
  const int m0 = blockIdx.x * 256 + wave * 32;
  const int n0 = blockIdx.y * 64;
  v8f acc00 = {}, acc01 = {}, acc02 = {}, acc03 = {};
  v8f acc10 = {}, acc11 = {}, acc12 = {}, acc13 = {};
  const _Float16* a0p = A + (long)m0 * K;
  const _Float16* a1p = A + (long)(m0 + 16) * K;
  const _Float16* b0p = Bw + (long)(n0 +  0) * K;
  const _Float16* b1p = Bw + (long)(n0 + 16) * K;
  const _Float16* b2p = Bw + (long)(n0 + 32) * K;
  const _Float16* b3p = Bw + (long)(n0 + 48) * K;
  for (int k0 = 0; k0 < K; k0 += 32) {
    v16h a0 = load_frag_a(a0p + k0, K, lane);
    v16h a1 = load_frag_a(a1p + k0, K, lane);
    v16h b0 = load_frag_b(b0p + k0, K, lane);
    v16h b1 = load_frag_b(b1p + k0, K, lane);
    v16h b2 = load_frag_b(b2p + k0, K, lane);
    v16h b3 = load_frag_b(b3p + k0, K, lane);
    acc00 = wmma_f16(a0, b0, acc00);
    acc01 = wmma_f16(a0, b1, acc01);
    acc02 = wmma_f16(a0, b2, acc02);
    acc03 = wmma_f16(a0, b3, acc03);
    acc10 = wmma_f16(a1, b0, acc10);
    acc11 = wmma_f16(a1, b1, acc11);
    acc12 = wmma_f16(a1, b2, acc12);
    acc13 = wmma_f16(a1, b3, acc13);
    // prefetch 8 k-steps (512B) ahead; speculative, OOB is silently dropped
    const int kp = k0 + 256;
    __builtin_prefetch(a0p + (lane & 15) * K + kp, 0, 1);
    __builtin_prefetch(a1p + (lane & 15) * K + kp, 0, 1);
    __builtin_prefetch(b0p + (lane & 15) * K + kp, 0, 1);
    __builtin_prefetch(b2p + (lane & 15) * K + kp, 0, 1);
  }
  const int rb  = ((lane >> 4) << 3);
  const int col = lane & 15;
  for (int r = 0; r < 8; ++r) {
    long row0 = m0 + rb + r;
    long row1 = row0 + 16;
    if (storeHalf) {
      Ch[row0 * N + n0 +      col] = (_Float16)acc00[r];
      Ch[row0 * N + n0 + 16 + col] = (_Float16)acc01[r];
      Ch[row0 * N + n0 + 32 + col] = (_Float16)acc02[r];
      Ch[row0 * N + n0 + 48 + col] = (_Float16)acc03[r];
      Ch[row1 * N + n0 +      col] = (_Float16)acc10[r];
      Ch[row1 * N + n0 + 16 + col] = (_Float16)acc11[r];
      Ch[row1 * N + n0 + 32 + col] = (_Float16)acc12[r];
      Ch[row1 * N + n0 + 48 + col] = (_Float16)acc13[r];
    } else {
      Cf[row0 * N + n0 +      col] = acc00[r];
      Cf[row0 * N + n0 + 16 + col] = acc01[r];
      Cf[row0 * N + n0 + 32 + col] = acc02[r];
      Cf[row0 * N + n0 + 48 + col] = acc03[r];
      Cf[row1 * N + n0 +      col] = acc10[r];
      Cf[row1 * N + n0 + 16 + col] = acc11[r];
      Cf[row1 * N + n0 + 32 + col] = acc12[r];
      Cf[row1 * N + n0 + 48 + col] = acc13[r];
    }
  }
}

// ---- QKV epilogue: RMS-norm(q,k), RoPE(q,k), v-mix; reshape ----------------
// qkv: [B*T, 3072] f16 (q|k|v). One wave per (b,t,h); lane owns (d, d+32)
// which is exactly the rotate_half pair.
__global__ __launch_bounds__(256)
void qkv_epilogue(const _Float16* __restrict__ qkv, const float* __restrict__ v1,
                  const float* __restrict__ cosb, const float* __restrict__ sinb,
                  const float* __restrict__ lamb_p,
                  _Float16* __restrict__ qg,   // [B,H,T,D]
                  _Float16* __restrict__ kg,   // [B,H,T,D]
                  _Float16* __restrict__ vtg) {// [B,H,D,T]
  const int lane = threadIdx.x & 31;
  const int wave = threadIdx.x >> 5;
  const long row = (long)blockIdx.x * 8 + wave;   // (b*T + t)*H + h
  const int  h  = (int)(row & (NHEAD - 1));
  const long bt = row >> 4;                       // b*T + t
  const int  t  = (int)(bt & (TLEN - 1));
  const int  b  = (int)(bt >> 12);
  const float lamb = *lamb_p;

  const _Float16* qr = qkv + bt * 3072 +            h * HDIM;
  const _Float16* kr = qkv + bt * 3072 + CDIM   +   h * HDIM;
  const _Float16* vr = qkv + bt * 3072 + 2*CDIM +   h * HDIM;
  const int d = lane;
  float q1 = (float)qr[d], q2 = (float)qr[d + 32];
  float k1 = (float)kr[d], k2 = (float)kr[d + 32];
  float va = (float)vr[d], vb = (float)vr[d + 32];

  float sq = q1 * q1 + q2 * q2;
  float sk = k1 * k1 + k2 * k2;
  for (int off = 16; off; off >>= 1) {
    sq += __shfl_xor(sq, off, 32);
    sk += __shfl_xor(sk, off, 32);
  }
  const float rq = rsqrtf(sq * (1.0f / 64.0f) + 1e-6f);
  const float rk = rsqrtf(sk * (1.0f / 64.0f) + 1e-6f);
  q1 *= rq; q2 *= rq; k1 *= rk; k2 *= rk;

  const float* cr = cosb + bt * HDIM;
  const float* sr = sinb + bt * HDIM;
  const float c1 = cr[d], c2 = cr[d + 32], s1 = sr[d], s2 = sr[d + 32];
  const float qo1 = q1 * c1 - q2 * s1, qo2 = q2 * c2 + q1 * s2;
  const float ko1 = k1 * c1 - k2 * s1, ko2 = k2 * c2 + k1 * s2;

  const float* v1r = v1 + bt * CDIM + h * HDIM;
  const float vm1 = (1.0f - lamb) * va + lamb * v1r[d];
  const float vm2 = (1.0f - lamb) * vb + lamb * v1r[d + 32];

  const long qi = (((long)b * NHEAD + h) * TLEN + t) * HDIM;
  qg[qi + d] = (_Float16)qo1; qg[qi + d + 32] = (_Float16)qo2;
  kg[qi + d] = (_Float16)ko1; kg[qi + d + 32] = (_Float16)ko2;
  const long vi = (((long)b * NHEAD + h) * HDIM) * TLEN + t;
  vtg[vi + (long)d * TLEN]        = (_Float16)vm1;
  vtg[vi + (long)(d + 32) * TLEN] = (_Float16)vm2;
}

// ---- Sliding-window attention: one WG per (block, b*h) ---------------------
// 8 waves x 16 query rows = 128 queries; 256-key window.
// LDS: spm f16 [128][256] (64KB, scores -> exp -> p in place)
//    | kw f16 [256][64] (32KB) | vt f16 [64][256] (32KB) | red f32[512] (2KB)
// => 130KB total -> two workgroups fit per 320KB WGP.
__global__ __launch_bounds__(256)
void swa_attn(const _Float16* __restrict__ qg, const _Float16* __restrict__ kg,
              const _Float16* __restrict__ vtg, _Float16* __restrict__ og) {
  extern __shared__ char smem[];
  _Float16* spm = (_Float16*)smem;                      // [128][256]
  _Float16* kw  = (_Float16*)(smem + 65536);            // [256][64]
  _Float16* vt  = (_Float16*)(smem + 65536 + 32768);    // [64][256]
  float*    red = (float*)(smem + 131072);              // [512]

  const int nb   = blockIdx.x;        // query block 0..31
  const int bh   = blockIdx.y;        // b*H + h
  const int tid  = threadIdx.x;
  const int lane = tid & 31;
  const int wave = tid >> 5;
  const int t0   = nb * WIN;

  const _Float16* kbase = kg  + (long)bh * TLEN * HDIM;
  const _Float16* vbase = vtg + (long)bh * HDIM * TLEN;

  // stage key window [256][64] (rows j<128 = previous block; zeros for nb==0)
  {
    const int j  = tid;
    const int tj = t0 - WIN + j;
    v8h* dst = (v8h*)(kw + j * HDIM);
    if (tj < 0) {
      v8h z = {};
      for (int i = 0; i < 8; ++i) dst[i] = z;
    } else {
      const v8h* src = (const v8h*)(kbase + (long)tj * HDIM);
      for (int i = 0; i < 8; ++i) dst[i] = src[i];
    }
  }
  // stage V^T window [64][256]
  {
    const int d = tid >> 2, part = tid & 3;
    const int jb = part * 64;
    const int tj = t0 - WIN + jb;
    v8h* dst = (v8h*)(vt + d * 256 + jb);
    if (tj < 0) {
      v8h z = {};
      for (int i = 0; i < 8; ++i) dst[i] = z;
    } else {
      const v8h* src = (const v8h*)(vbase + (long)d * TLEN + tj);
      for (int i = 0; i < 8; ++i) dst[i] = src[i];
    }
  }
  __syncthreads();

  // scores = q @ kw^T * 1/sqrt(D), masked, into LDS (f16)
  const int m0 = wave * 16;
  const _Float16* qrow = qg + ((long)bh * TLEN + t0 + m0) * HDIM;
  for (int jt = 0; jt < 16; ++jt) {
    v8f acc = {};
    for (int ks = 0; ks < 2; ++ks) {
      v16h a = load_frag_a(qrow + ks * 32, HDIM, lane);
      v16h b = load_frag_b(kw + (jt * 16) * HDIM + ks * 32, HDIM, lane);
      acc = wmma_f16(a, b, acc);
    }
    const int ib = m0 + ((lane >> 4) << 3);
    const int j  = jt * 16 + (lane & 15);
    for (int r = 0; r < 8; ++r) {
      const int ti = ib + r;
      const int diff = ti - j + WIN;
      const bool valid = (diff >= 0) && (diff < WIN) && ((nb > 0) || (j >= WIN));
      spm[ti * 256 + j] = valid ? (_Float16)(acc[r] * 0.125f) : (_Float16)(-30000.0f);
    }
  }
  __syncthreads();

  // softmax: 2 threads per row (128 cols each); f32 max/sum via LDS scratch
  {
    const int row = tid & 127, half = tid >> 7;
    _Float16* rp = spm + row * 256 + half * 128;
    float mx = -1e30f;
    for (int j = 0; j < 128; ++j) mx = fmaxf(mx, (float)rp[j]);
    red[half * 128 + row] = mx;
    __syncthreads();
    const float m = fmaxf(red[row], red[128 + row]);
    float s = 0.0f;
    for (int j = 0; j < 128; ++j) {
      const float e = __expf((float)rp[j] - m);
      rp[j] = (_Float16)e;          // own slots only; race-free
      s += e;
    }
    red[256 + half * 128 + row] = s;
    __syncthreads();
    const float inv = 1.0f / (red[256 + row] + red[384 + row]);
    for (int j = 0; j < 128; ++j) rp[j] = (_Float16)((float)rp[j] * inv);
  }
  __syncthreads();

  // out = p @ v : each wave its 16x64 tile -> [B,T,H,D] f16
  const int b = bh >> 4, h = bh & (NHEAD - 1);
  for (int dt = 0; dt < 4; ++dt) {
    v8f acc = {};
    for (int ks = 0; ks < 8; ++ks) {
      v16h a   = load_frag_a(spm + m0 * 256 + ks * 32, 256, lane);
      v16h bfr = load_frag_b(vt + (dt * 16) * 256 + ks * 32, 256, lane);
      acc = wmma_f16(a, bfr, acc);
    }
    const int rbase = m0 + ((lane >> 4) << 3);
    const int col   = dt * 16 + (lane & 15);
    for (int r = 0; r < 8; ++r) {
      const int t = t0 + rbase + r;
      og[(((long)b * TLEN + t) * NHEAD + h) * HDIM + col] = (_Float16)acc[r];
    }
  }
}

// ---------------------------------------------------------------------------
extern "C" void kernel_launch(void* const* d_in, const int* in_sizes, int n_in,
                              void* d_out, int out_size, void* d_ws, size_t ws_size,
                              hipStream_t stream) {
  (void)in_sizes; (void)n_in; (void)out_size; (void)ws_size;
  const float* x     = (const float*)d_in[0];   // [B,T,C]
  const float* v1    = (const float*)d_in[1];   // [B,T,H,D]
  const float* cosb  = (const float*)d_in[2];   // [B,T,D]
  const float* sinb  = (const float*)d_in[3];   // [B,T,D]
  const float* Wq    = (const float*)d_in[4];   // [C,C]
  const float* Wk    = (const float*)d_in[5];
  const float* Wv    = (const float*)d_in[6];
  const float* Wproj = (const float*)d_in[7];
  const float* lamb  = (const float*)d_in[8];

  const long M  = (long)BATCH * TLEN;           // 16384
  const long CC = (long)CDIM * CDIM;            // 1,048,576

  // workspace carve-up (f16 intermediates), ~264 MB total
  char* ws = (char*)d_ws;
  _Float16* xh   = (_Float16*)ws;                           ws += M * CDIM * 2;       // 32MB
  _Float16* Wcat = (_Float16*)ws;                           ws += 3 * CC * 2;         // 6MB
  _Float16* Wph  = (_Float16*)ws;                           ws += CC * 2;             // 2MB
  _Float16* qkvh = (_Float16*)ws;                           ws += M * 3 * CDIM * 2;   // 96MB
  _Float16* qg   = (_Float16*)ws;                           ws += M * CDIM * 2;       // 32MB
  _Float16* kg   = (_Float16*)ws;                           ws += M * CDIM * 2;       // 32MB
  _Float16* vtg  = (_Float16*)ws;                           ws += M * CDIM * 2;       // 32MB
  _Float16* ah   = (_Float16*)ws;                           ws += M * CDIM * 2;       // 32MB

  // 1) fp32 -> f16 conversions (8 elems/thread)
  {
    long n8 = (M * CDIM) / 8;
    cvt_f32_f16_v8<<<dim3((unsigned)((n8 + 255) / 256)), dim3(256), 0, stream>>>(
        (const float4*)x, (v8h*)xh, n8);
    long w8 = CC / 8;
    unsigned wg = (unsigned)((w8 + 255) / 256);
    cvt_f32_f16_v8<<<dim3(wg), dim3(256), 0, stream>>>((const float4*)Wq, (v8h*)Wcat, w8);
    cvt_f32_f16_v8<<<dim3(wg), dim3(256), 0, stream>>>((const float4*)Wk, (v8h*)(Wcat + CC), w8);
    cvt_f32_f16_v8<<<dim3(wg), dim3(256), 0, stream>>>((const float4*)Wv, (v8h*)(Wcat + 2 * CC), w8);
    cvt_f32_f16_v8<<<dim3(wg), dim3(256), 0, stream>>>((const float4*)Wproj, (v8h*)Wph, w8);
  }

  // 2) fused QKV GEMM: [16384,1024] x [3072,1024]^T -> f16 [16384,3072]
  gemm_nt_f16<<<dim3(64, 48), dim3(256), 0, stream>>>(
      xh, Wcat, nullptr, qkvh, (int)M, 3 * CDIM, CDIM, 1);

  // 3) RMS + RoPE + v-mix epilogue, reshape to attention-friendly layouts
  qkv_epilogue<<<dim3(32768), dim3(256), 0, stream>>>(
      qkvh, v1, cosb, sinb, lamb, qg, kg, vtg);

  // 4) sliding-window attention (130KB dynamic LDS per WG)
  swa_attn<<<dim3(TLEN / WIN, BATCH * NHEAD), dim3(256), 133120, stream>>>(
      qg, kg, vtg, ah);

  // 5) output projection straight into d_out (fp32)
  gemm_nt_f16<<<dim3(64, 16), dim3(256), 0, stream>>>(
      ah, Wph, (float*)d_out, nullptr, (int)M, CDIM, CDIM, 0);

  // 6) second tuple output: v1 passthrough
  hipMemcpyAsync((float*)d_out + M * CDIM, v1, (size_t)M * CDIM * sizeof(float),
                 hipMemcpyDeviceToDevice, stream);
}
